// NutriGraphNetDecoder_11098195493028
// MI455X (gfx1250) — matrix operand order repaired and checked by reading, not compiled, not run on hardware
//
#include <hip/hip_runtime.h>

// ---------------------------------------------------------------------------
// NutriGraphNet decoder for MI455X (gfx1250, wave32, WMMA).
//
// Embeddings (77MB) are L2-resident (192MB L2) -> RECOMPUTE the [E,128]
// hidden activations in two passes instead of spilling 512MB to HBM.
// GEMMs run on v_wmma_f32_16x16x32_bf16 with a hi/lo bf16 split (3 WMMAs
// per tile) for ~fp32 accuracy.  b1 cancels under BatchNorm; mean/gamma/beta
// fold into per-column scale/shift.  BN stats reduced deterministically.
//
// v4: fix async builtin arg types -- the diagnostic revealed param0 is
// "int __vector(4) __device__ *" (global-AS v4i*), so cast both args to
// exact vector-pointer types.  Dual accumulators for WMMA ILP; memory
// clobbers keep LICM from hoisting LDS B-fragments out of the edge loop.
// ---------------------------------------------------------------------------

typedef __bf16 bf16_t;
typedef bf16_t v16bf __attribute__((ext_vector_type(16)));
typedef float  v8f   __attribute__((ext_vector_type(8)));
typedef int    v4i_  __attribute__((vector_size(16)));

#define AS1 __attribute__((address_space(1)))
#define AS3 __attribute__((address_space(3)))

#define Hdim   128
#define K2     256
#define NB1    512              // pass-1 grid (fixed -> deterministic partials)
#define NB2    512
#define WAVES  8
#define BLOCK_EDGES (WAVES * 16)

__device__ __forceinline__ bf16_t f2bf(float x) { return (bf16_t)x; }

// ---------------------------------------------------------------------------
// Stage 128KB of W1^T (hi|lo) into LDS.  Prefer the CDNA5 async DMA path
// (GLOBAL_LOAD_ASYNC_TO_LDS_B128, ASYNCcnt-tracked, no VGPR round-trip).
// ---------------------------------------------------------------------------
__device__ __forceinline__ void stage_w1(uint4* lB, const uint4* sh,
                                         const uint4* sl, int tid) {
#if defined(__gfx1250__) && __has_builtin(__builtin_amdgcn_global_load_async_to_lds_b128)
#pragma unroll
  for (int i = 0; i < 16; ++i) {
    __builtin_amdgcn_global_load_async_to_lds_b128(
        (AS1 v4i_*)(sh + tid + i * 256),
        (AS3 v4i_*)(lB + tid + i * 256), 0, 0);
    __builtin_amdgcn_global_load_async_to_lds_b128(
        (AS1 v4i_*)(sl + tid + i * 256),
        (AS3 v4i_*)(lB + 4096 + tid + i * 256), 0, 0);
  }
#if __has_builtin(__builtin_amdgcn_s_wait_asynccnt)
  __builtin_amdgcn_s_wait_asynccnt(0);
#else
  asm volatile("s_wait_asynccnt 0" ::: "memory");
#endif
#else
#pragma unroll
  for (int i = 0; i < 16; ++i) {
    lB[tid + i * 256]        = sh[tid + i * 256];
    lB[4096 + tid + i * 256] = sl[tid + i * 256];
  }
#endif
}

// A-fragment for one K-step s (K = s*32 .. s*32+31) of the concat row.
// ISA 16-bit A layout: lanes 0-15 hold K = {khalf..khalf+7, 16+khalf..}, with
// khalf = 8*(lane>>4).  hi/lo bf16 split for accuracy.
__device__ __forceinline__ void load_a_frag(const float* __restrict__ uptr,
                                            const float* __restrict__ fptr,
                                            int khalf, int s,
                                            v16bf& ah, v16bf& al) {
#pragma unroll
  for (int c = 0; c < 2; ++c) {
    const int kk = s * 32 + c * 16 + khalf;           // multiple of 8, no straddle
    const float* p = (kk < Hdim) ? (uptr + kk) : (fptr + (kk - Hdim));
    const float4 q0 = ((const float4*)p)[0];
    const float4 q1 = ((const float4*)p)[1];
    const float xs[8] = {q0.x, q0.y, q0.z, q0.w, q1.x, q1.y, q1.z, q1.w};
#pragma unroll
    for (int j = 0; j < 8; ++j) {
      const float x = xs[j];
      const bf16_t h = f2bf(x);
      ah[c * 8 + j] = h;
      al[c * 8 + j] = f2bf(x - (float)h);
    }
  }
}

// B-fragment from LDS: row n of W1^T (256 bf16), half-lane takes K = {8h..8h+7, 16+8h..16+8h+7}.
__device__ __forceinline__ v16bf load_b_frag(const uint4* __restrict__ lB,
                                             int n, int s, int half) {
  const uint4* bp = lB + (n * 32 + s * 4 + half);
  struct U8 { uint4 a, b; } u;
  u.a = bp[0];
  u.b = bp[2];
  return __builtin_bit_cast(v16bf, u);
}

// 16x16 tile MACC: acc0 += Ahi*Bhi ; acc1 += Ahi*Blo + Alo*Bhi  (2-way ILP).
__device__ __forceinline__ void tile_macc(const uint4* lB, int n, int half,
                                          const v16bf* ah, const v16bf* al,
                                          v8f& acc0, v8f& acc1) {
#pragma unroll
  for (int s = 0; s < 8; ++s) {
    const v16bf bh = load_b_frag(lB,        n, s, half);
    const v16bf bl = load_b_frag(lB + 4096, n, s, half);
    acc0 = __builtin_amdgcn_wmma_f32_16x16x32_bf16(false, ah[s], false, bh,
                                                   (short)0, acc0, false, false);
    acc1 = __builtin_amdgcn_wmma_f32_16x16x32_bf16(false, ah[s], false, bl,
                                                   (short)0, acc1, false, false);
    acc1 = __builtin_amdgcn_wmma_f32_16x16x32_bf16(false, al[s], false, bh,
                                                   (short)0, acc1, false, false);
  }
}

// ---------------------------------------------------------------------------
// Prep: W1 [256][128] f32 row-major  ->  W1^T hi/lo bf16 [128][256] in ws.
// ---------------------------------------------------------------------------
__global__ void prep_kernel(const float* __restrict__ W1,
                            bf16_t* __restrict__ w1hi,
                            bf16_t* __restrict__ w1lo) {
  const int idx = blockIdx.x * 256 + threadIdx.x;    // n*256 + k
  if (idx < Hdim * K2) {
    const int n = idx >> 8;
    const int k = idx & 255;
    const float x = W1[k * Hdim + n];
    const bf16_t h = f2bf(x);
    w1hi[idx] = h;
    w1lo[idx] = f2bf(x - (float)h);
  }
}

// ---------------------------------------------------------------------------
// Pass 1: z = concat(user,food) @ W1 ; accumulate per-column sum / sumsq.
// Deterministic: regs -> LDS tree -> per-block partials (fixed slots).
// ---------------------------------------------------------------------------
__global__ __launch_bounds__(256, 1) void pass1_kernel(
    const float* __restrict__ user_emb, const float* __restrict__ food_emb,
    const int* __restrict__ row, const int* __restrict__ col,
    const bf16_t* __restrict__ w1hi, const bf16_t* __restrict__ w1lo,
    float* __restrict__ blkpart, int E) {
  __shared__ uint4 lB[8192];                 // 128KB: [0..4095]=hi, [4096..]=lo
  const int tid = threadIdx.x;
  stage_w1(lB, (const uint4*)w1hi, (const uint4*)w1lo, tid);
  __syncthreads();

  const int lane  = tid & 31;
  const int wave  = tid >> 5;
  const int nloc  = lane & 15;
  const int half  = lane >> 4;
  const int khalf = half * 8;

  float cs[8], cq[8];
#pragma unroll
  for (int t = 0; t < 8; ++t) { cs[t] = 0.f; cq[t] = 0.f; }

  for (int base0 = blockIdx.x * BLOCK_EDGES; base0 < E; base0 += NB1 * BLOCK_EDGES) {
    const int base = base0 + wave * 16;
    const int m  = base + nloc;
    const int mc = (m < E) ? m : (E - 1);
    const float* uptr = user_emb + (long long)row[mc] * Hdim;
    const float* fptr = food_emb + (long long)col[mc] * Hdim;

    // A-fragments for all 8 K-steps stay resident (128 VGPRs, by design).
    v16bf ah[8], al[8];
#pragma unroll
    for (int s = 0; s < 8; ++s) load_a_frag(uptr, fptr, khalf, s, ah[s], al[s]);

#pragma unroll
    for (int t = 0; t < 8; ++t) {
      // Opaque memory clobber: stops LICM from hoisting the loop-invariant
      // LDS B-fragments out of the edge loop.
      asm volatile("" ::: "memory");
      const v8f vzero = {0.f, 0.f, 0.f, 0.f, 0.f, 0.f, 0.f, 0.f};
      v8f acc0 = vzero, acc1 = vzero;
      tile_macc(lB, t * 16 + nloc, half, ah, al, acc0, acc1);
      // D layout: lane<16 -> rows base+0..7, lane>=16 -> rows base+8..15.
#pragma unroll
      for (int r = 0; r < 8; ++r) {
        const int mm = base + half * 8 + r;
        if (mm < E) {
          const float z = acc0[r] + acc1[r];
          cs[t] += z;
          cq[t] += z * z;
        }
      }
    }
  }

  // Block reduction (reuse lB as scratch) in a fixed order.
  __syncthreads();
  float* red = (float*)lB;                       // [WAVES][256]
#pragma unroll
  for (int t = 0; t < 8; ++t) {
    const float s = cs[t] + __shfl_xor(cs[t], 16);
    const float q = cq[t] + __shfl_xor(cq[t], 16);
    if (half == 0) {
      red[wave * 256 + t * 16 + nloc]       = s;
      red[wave * 256 + 128 + t * 16 + nloc] = q;
    }
  }
  __syncthreads();
  float a = 0.f;
#pragma unroll
  for (int w = 0; w < WAVES; ++w) a += red[w * 256 + tid];
  blkpart[blockIdx.x * 256 + tid] = a;
}

// ---------------------------------------------------------------------------
// Finalize BN: scale = gamma*rsqrt(var+eps), shift = beta - mean*scale.
// (b1 cancels under BatchNorm.)
// ---------------------------------------------------------------------------
__global__ void stats_kernel(const float* __restrict__ blkpart,
                             const float* __restrict__ gamma_,
                             const float* __restrict__ beta_,
                             float* __restrict__ scale,
                             float* __restrict__ shift, int E) {
  const int n = threadIdx.x;                     // 128 threads
  float s = 0.f, q = 0.f;
  for (int b = 0; b < NB1; ++b) {
    s += blkpart[b * 256 + n];
    q += blkpart[b * 256 + 128 + n];
  }
  const float invE = 1.0f / (float)E;
  const float mean = s * invE;
  const float var  = q * invE - mean * mean;
  const float sc   = gamma_[n] * rsqrtf(var + 1e-5f);
  scale[n] = sc;
  shift[n] = beta_[n] - mean * sc;
}

// ---------------------------------------------------------------------------
// Pass 2: recompute z (L2-resident gathers), fused BN -> ReLU -> dot(W2) ->
// sigmoid.  Cross-lane dot via one xor-butterfly (linearity of reduction).
// ---------------------------------------------------------------------------
__global__ __launch_bounds__(256, 1) void pass2_kernel(
    const float* __restrict__ user_emb, const float* __restrict__ food_emb,
    const int* __restrict__ row, const int* __restrict__ col,
    const bf16_t* __restrict__ w1hi, const bf16_t* __restrict__ w1lo,
    const float* __restrict__ scale, const float* __restrict__ shift,
    const float* __restrict__ W2, const float* __restrict__ b2,
    float* __restrict__ out, int E) {
  __shared__ uint4 lB[8192];
  const int tid = threadIdx.x;
  stage_w1(lB, (const uint4*)w1hi, (const uint4*)w1lo, tid);
  __syncthreads();

  const int lane  = tid & 31;
  const int wave  = tid >> 5;
  const int nloc  = lane & 15;
  const int half  = lane >> 4;
  const int khalf = half * 8;

  float scr[8], shr[8], w2r[8];
#pragma unroll
  for (int t = 0; t < 8; ++t) {
    const int n = t * 16 + nloc;
    scr[t] = scale[n];
    shr[t] = shift[n];
    w2r[t] = W2[n];
  }
  const float b2v = b2[0];

  for (int base0 = blockIdx.x * BLOCK_EDGES; base0 < E; base0 += NB2 * BLOCK_EDGES) {
    const int base = base0 + wave * 16;
    const int m  = base + nloc;
    const int mc = (m < E) ? m : (E - 1);
    const float* uptr = user_emb + (long long)row[mc] * Hdim;
    const float* fptr = food_emb + (long long)col[mc] * Hdim;

    v16bf ah[8], al[8];
#pragma unroll
    for (int s = 0; s < 8; ++s) load_a_frag(uptr, fptr, khalf, s, ah[s], al[s]);

    float pAcc[8];
#pragma unroll
    for (int r = 0; r < 8; ++r) pAcc[r] = 0.f;

#pragma unroll
    for (int t = 0; t < 8; ++t) {
      asm volatile("" ::: "memory");           // block LICM of LDS B-frags
      const v8f vzero = {0.f, 0.f, 0.f, 0.f, 0.f, 0.f, 0.f, 0.f};
      v8f acc0 = vzero, acc1 = vzero;
      tile_macc(lB, t * 16 + nloc, half, ah, al, acc0, acc1);
      // Fused epilogue for this tile: y = relu(z*scale+shift); dot with W2.
#pragma unroll
      for (int r = 0; r < 8; ++r) {
        float y = (acc0[r] + acc1[r]) * scr[t] + shr[t];
        y = (y > 0.f) ? y : 0.f;
        pAcc[r] += y * w2r[t];
      }
    }

    // One 16-lane butterfly completes the 128-wide dot per edge row.
#pragma unroll
    for (int r = 0; r < 8; ++r) {
#pragma unroll
      for (int off = 1; off < 16; off <<= 1)
        pAcc[r] += __shfl_xor(pAcc[r], off);
    }
#pragma unroll
    for (int r = 0; r < 8; ++r) {
      const int mm = base + half * 8 + r;
      if (nloc == r && mm < E) {
        const float x = pAcc[r] + b2v;
        out[mm] = 1.0f / (1.0f + __expf(-x));
      }
    }
  }
}

// ---------------------------------------------------------------------------
extern "C" void kernel_launch(void* const* d_in, const int* in_sizes, int n_in,
                              void* d_out, int out_size, void* d_ws, size_t ws_size,
                              hipStream_t stream) {
  const float* user_emb = (const float*)d_in[0];
  const float* food_emb = (const float*)d_in[1];
  const float* W1       = (const float*)d_in[2];
  // d_in[3] = b1 : cancels under BatchNorm, unused.
  const float* gamma_   = (const float*)d_in[4];
  const float* beta_    = (const float*)d_in[5];
  const float* W2       = (const float*)d_in[6];
  const float* b2       = (const float*)d_in[7];
  const int*   row      = (const int*)d_in[8];
  const int*   col      = (const int*)d_in[9];
  float*       out      = (float*)d_out;
  const int    E        = in_sizes[8];

  char* ws = (char*)d_ws;
  bf16_t* w1hi   = (bf16_t*)(ws);                          // 64 KB
  bf16_t* w1lo   = (bf16_t*)(ws + 65536);                  // 64 KB
  float*  blkpart= (float*)(ws + 131072);                  // NB1*256*4 = 512 KB
  float*  scale  = (float*)(ws + 131072 + NB1 * 256 * 4);  // 512 B
  float*  shift  = scale + Hdim;                           // 512 B

  prep_kernel<<<(Hdim * K2 + 255) / 256, 256, 0, stream>>>(W1, w1hi, w1lo);
  pass1_kernel<<<NB1, 256, 0, stream>>>(user_emb, food_emb, row, col,
                                        w1hi, w1lo, blkpart, E);
  stats_kernel<<<1, 128, 0, stream>>>(blkpart, gamma_, beta_, scale, shift, E);
  pass2_kernel<<<NB2, 256, 0, stream>>>(user_emb, food_emb, row, col,
                                        w1hi, w1lo, scale, shift, W2, b2, out, E);
}
